// MLPResNetBlockPro_85770496901585
// MI455X (gfx1250) — matrix operand order repaired and checked by reading, not compiled, not run on hardware
//
#include <hip/hip_runtime.h>
#include <hip/hip_bf16.h>

// ---------------------------------------------------------------------------
// Types for CDNA5 WMMA (wave32): v_wmma_f32_16x16x32_bf16
// ---------------------------------------------------------------------------
typedef __bf16 bf16_t;
typedef __attribute__((ext_vector_type(16))) __bf16 v16bf;
typedef __attribute__((ext_vector_type(8)))  float  v8f;

union FragB { v16bf v; uint4 u[2]; };

static __device__ __forceinline__ v8f wmma_bf16(v16bf a, v16bf b, v8f c) {
  // (neg_a, A, neg_b, B, c_mod, C, reuse_a, reuse_b)
  return __builtin_amdgcn_wmma_f32_16x16x32_bf16(false, a, false, b, (short)0, c,
                                                 false, false);
}

// CDNA5 async global->LDS copy (ASYNCcnt tracked). VDST = LDS byte offset
// (AS3 pointer, 32-bit VGPR), VADDR = 64-bit global address.
typedef __attribute__((address_space(3))) void lds_void_t;
static __device__ __forceinline__ void async_copy_b128(lds_void_t* lds,
                                                       const void* gl) {
  asm volatile("global_load_async_to_lds_b128 %0, %1, off"
               :: "v"(lds), "v"(gl) : "memory");
}
static __device__ __forceinline__ void wait_async0() {
  asm volatile("s_wait_asynccnt 0x0" ::: "memory");
}

// Problem constants
#define Bc    4
#define Tc    1024
#define DIMc  1024
#define Hc    16
#define HDc   64
#define SRCc  2048
#define KAc   320      /* 256 adapter + 64 prompt */
#define KTc   256
#define Sall  1600     /* 1024 + 320 + 256 */
#define OFF_A 1024
#define OFF_T 1344

// ---------------------------------------------------------------------------
// fp32 -> bf16 convert
// ---------------------------------------------------------------------------
__global__ __launch_bounds__(256) void cvt_bf16_kernel(const float* __restrict__ s,
                                                       bf16_t* __restrict__ d, int n) {
  int id = blockIdx.x * 256 + threadIdx.x;
  if (id < n) d[id] = (bf16_t)s[id];
}

// Build bf16 h_adapter = concat(h_a, p) along seq axis (per batch)
__global__ __launch_bounds__(256) void build_adapter_kernel(const float* __restrict__ h_a,
                                                            const float* __restrict__ p,
                                                            bf16_t* __restrict__ dst) {
  int id = blockIdx.x * 256 + threadIdx.x;            // B*320*2048 threads
  int c  = id & (SRCc - 1);
  int r  = id >> 11;                                  // / 2048
  int b  = r / KAc;
  int ri = r - b * KAc;
  float v = (ri < 256) ? h_a[((size_t)(b * 256 + ri)) * SRCc + c]
                       : p[((size_t)(b * 64 + (ri - 256))) * SRCc + c];
  dst[id] = (bf16_t)v;
}

// Transpose + convert: W (K x N, fp32) -> Wt (N x K, bf16)
__global__ __launch_bounds__(256) void transpose_cvt_kernel(const float* __restrict__ W,
                                                            bf16_t* __restrict__ Wt,
                                                            int K, int N) {
  __shared__ float tile[32][33];
  int n0 = blockIdx.x * 32, k0 = blockIdx.y * 32;
  int tx = threadIdx.x, ty = threadIdx.y;             // block (32, 8)
  for (int i = 0; i < 32; i += 8)
    tile[ty + i][tx] = W[(size_t)(k0 + ty + i) * N + (n0 + tx)];
  __syncthreads();
  for (int i = 0; i < 32; i += 8)
    Wt[(size_t)(n0 + ty + i) * K + (k0 + tx)] = (bf16_t)tile[tx][ty + i];
}

// ---------------------------------------------------------------------------
// Generic WMMA GEMM:  C[M,N] (fp32) = A[M,K] (bf16, row-major) @ Wt[N,K]^T
// Block = 8 waves computing 128(M) x 64(N); the shared 64x32 B tile is staged
// into LDS once per K-chunk with async global->LDS copies (double-buffered,
// ASYNCcnt + barrier), then read back as B-fragments via ds_load_b128.
// Requires M%128==0, N%64==0, K%32==0 (true for every call here).
// ---------------------------------------------------------------------------
__global__ __launch_bounds__(256) void gemm_bf16_wmma_kernel(const bf16_t* __restrict__ A,
                                                             const bf16_t* __restrict__ Wt,
                                                             float* __restrict__ C,
                                                             int M, int N, int K) {
  __shared__ __align__(16) bf16_t sB[2][64 * 32];
  const int tid = threadIdx.x;
  const int wave = tid >> 5;
  const int lane = tid & 31;
  const int ln = lane & 15, lh = lane >> 4;
  const int m0 = blockIdx.y * 128 + wave * 16;
  const int n0 = blockIdx.x * 64;
  (void)M;

  // One 16B async transfer per thread per K-chunk: 256 threads x 16B = 4KB tile
  const int bn = tid >> 2;                 // B-tile row (N-local), 0..63
  const int bk = (tid & 3) * 8;            // K offset within 32-chunk
  const bf16_t* gB = Wt + (size_t)(n0 + bn) * K + bk;
  lds_void_t* lbuf0 = (lds_void_t*)&sB[0][bn * 32 + bk];
  lds_void_t* lbuf1 = (lds_void_t*)&sB[1][bn * 32 + bk];

  const bf16_t* arow = A + (size_t)(m0 + ln) * K;
  v8f acc0 = {}, acc1 = {}, acc2 = {}, acc3 = {};

  async_copy_b128(lbuf0, gB);              // prologue: stage chunk 0
  const int nk = K / 32;
  for (int i = 0; i < nk; ++i) {
    wait_async0();                         // this wave's issued copies done
    __syncthreads();                       // all waves' copies published
    if (i + 1 < nk)                        // prefetch next chunk into other buf
      async_copy_b128((i & 1) ? lbuf0 : lbuf1, gB + (i + 1) * 32);

    const int k = i * 32;
    FragB a;
    a.u[0] = *(const uint4*)(arow + k + lh * 16);      // K chunk lh*16 .. +7
    a.u[1] = *(const uint4*)(arow + k + lh * 16 + 8);  // K chunk +8
    __builtin_prefetch(arow + k + 256, 0, 0);          // global_prefetch_b8

    const bf16_t* sb = sB[i & 1];
    const bf16_t* s0 = sb + (0 * 16 + ln) * 32 + lh * 16;
    const bf16_t* s1 = sb + (1 * 16 + ln) * 32 + lh * 16;
    const bf16_t* s2 = sb + (2 * 16 + ln) * 32 + lh * 16;
    const bf16_t* s3 = sb + (3 * 16 + ln) * 32 + lh * 16;
    FragB b0, b1, b2, b3;
    b0.u[0] = *(const uint4*)(s0); b0.u[1] = *(const uint4*)(s0 + 8);
    b1.u[0] = *(const uint4*)(s1); b1.u[1] = *(const uint4*)(s1 + 8);
    b2.u[0] = *(const uint4*)(s2); b2.u[1] = *(const uint4*)(s2 + 8);
    b3.u[0] = *(const uint4*)(s3); b3.u[1] = *(const uint4*)(s3 + 8);

    acc0 = wmma_bf16(a.v, b0.v, acc0);
    acc1 = wmma_bf16(a.v, b1.v, acc1);
    acc2 = wmma_bf16(a.v, b2.v, acc2);
    acc3 = wmma_bf16(a.v, b3.v, acc3);

    __syncthreads();                       // reads done before buf reuse
  }

  // D layout: row = lh*8 + v, col = ln
  for (int v = 0; v < 8; ++v) {
    size_t row = (size_t)(m0 + lh * 8 + v) * N + n0;
    C[row + 0 * 16 + ln] = acc0[v];
    C[row + 1 * 16 + ln] = acc1[v];
    C[row + 2 * 16 + ln] = acc2[v];
    C[row + 3 * 16 + ln] = acc3[v];
  }
}

// ---------------------------------------------------------------------------
// Bias + (optional RoPE) + scale, scatter into attention layouts (bf16).
//   transposed==0: dst[((b*H+h)*Sdst + s_off + s_in)*64 + d]   (q / k)
//   transposed==1: dst[((b*H+h)*64 + d)*Sdst + s_off + s_in]   (v, Sdst=1600)
// RoPE matches reference: angle = pos * base^(-(d&31)/32); interleaved rotate.
// ---------------------------------------------------------------------------
__global__ __launch_bounds__(256) void arrange_kernel(const float* __restrict__ Craw,
                                                      const float* __restrict__ bias,
                                                      bf16_t* __restrict__ dst,
                                                      int rpb, int Sdst, int s_off,
                                                      int transposed, int do_rope,
                                                      float scale,
                                                      const float* __restrict__ gptr,
                                                      int ntot) {
  int id = blockIdx.x * 256 + threadIdx.x;
  if (id >= ntot) return;
  int c = id & (DIMc - 1);
  int r = id >> 10;
  int b = r / rpb;
  int s_in = r - b * rpb;
  int h = c >> 6, d = c & 63;

  float val = Craw[(size_t)r * DIMc + c] + bias[c];
  if (do_rope) {
    float partner = Craw[(size_t)r * DIMc + (c ^ 1)] + bias[c ^ 1];
    float rot = (d & 1) ? partner : -partner;   // even: -x[j+1], odd: +x[j-1]
    float ang = (float)s_in * __powf(10000.0f, -(float)(d & 31) / 32.0f);
    val = val * __cosf(ang) + rot * __sinf(ang);
  }
  float sc = scale;
  if (gptr) sc *= tanhf(gptr[0]);
  val *= sc;

  size_t idx;
  if (transposed)
    idx = ((size_t)(b * Hc + h) * 64 + d) * (size_t)Sdst + (s_off + s_in);
  else
    idx = ((size_t)(b * Hc + h) * (size_t)Sdst + (s_off + s_in)) * 64 + d;
  dst[idx] = (bf16_t)val;
}

// ---------------------------------------------------------------------------
// Flash attention, one wave per (b, h, 16-query tile).
// Scores computed transposed: S^T(16 keys x 16 q) = K_tile(A) x q^T(B),
// so each lane owns one query column -> softmax needs only shfl_xor(16).
// Output accumulated transposed: O^T(16 dims x 16 q) = V^T(A) x P^T(B).
// q pre-scaled by 1/8; k_t pre-scaled by tanh(g).
// ---------------------------------------------------------------------------
__global__ __launch_bounds__(256) void attn_kernel(const bf16_t* __restrict__ qh,
                                                   const bf16_t* __restrict__ kall,
                                                   const bf16_t* __restrict__ vT,
                                                   bf16_t* __restrict__ attn) {
  const int wave = threadIdx.x >> 5;
  const int lane = threadIdx.x & 31;
  const int ln = lane & 15, lh = lane >> 4;
  const int job = blockIdx.x * 8 + wave;      // B*H*(T/16) = 4096 jobs
  const int b = job >> 10;
  const int rem = job & 1023;
  const int h = rem >> 6;
  const int qt = rem & 63;

  const bf16_t* qrow = qh + ((size_t)(b * Hc + h) * Tc + (qt * 16 + ln)) * 64;
  const bf16_t* kb   = kall + (size_t)(b * Hc + h) * Sall * 64;
  const bf16_t* vb   = vT   + (size_t)(b * Hc + h) * 64 * Sall;

  // q^T B-fragments (fixed): lane = query column, dims lh*8..+7 and lh*8+16..+7
  FragB bq0, bq1;
  bq0.u[0] = *(const uint4*)(qrow + lh * 8);
  bq0.u[1] = *(const uint4*)(qrow + lh * 8 + 16);
  bq1.u[0] = *(const uint4*)(qrow + 32 + lh * 8);
  bq1.u[1] = *(const uint4*)(qrow + 32 + lh * 8 + 16);

  v8f o0 = {}, o1 = {}, o2 = {}, o3 = {};
  float mrun = -1e30f, lrun = 0.0f;

  for (int s = 0; s < Sall; s += 32) {
    // --- score tiles S^T (two 16-key tiles, K-dim = 64 -> 2 chained WMMA) ---
    v8f st0 = {}, st1 = {};
    {
      const bf16_t* kr = kb + (size_t)(s + ln) * 64;
      FragB a0, a1;
      a0.u[0] = *(const uint4*)(kr + lh * 8);
      a0.u[1] = *(const uint4*)(kr + lh * 8 + 16);
      a1.u[0] = *(const uint4*)(kr + 32 + lh * 8);
      a1.u[1] = *(const uint4*)(kr + 32 + lh * 8 + 16);
      st0 = wmma_bf16(a0.v, bq0.v, st0);
      st0 = wmma_bf16(a1.v, bq1.v, st0);
    }
    {
      const bf16_t* kr = kb + (size_t)(s + 16 + ln) * 64;
      FragB a0, a1;
      a0.u[0] = *(const uint4*)(kr + lh * 8);
      a0.u[1] = *(const uint4*)(kr + lh * 8 + 16);
      a1.u[0] = *(const uint4*)(kr + 32 + lh * 8);
      a1.u[1] = *(const uint4*)(kr + 32 + lh * 8 + 16);
      st1 = wmma_bf16(a0.v, bq0.v, st1);
      st1 = wmma_bf16(a1.v, bq1.v, st1);
    }

    // --- online softmax (lane owns query ln; partner lane^16 has other keys)
    float cmax = -1e30f;
    for (int v = 0; v < 8; ++v) {
      cmax = fmaxf(cmax, st0[v]);
      cmax = fmaxf(cmax, st1[v]);
    }
    cmax = fmaxf(cmax, __shfl_xor(cmax, 16, 32));
    float mnew = fmaxf(mrun, cmax);
    float f = __expf(mrun - mnew);
    float p0[8], p1[8];
    float csum = 0.0f;
    for (int v = 0; v < 8; ++v) {
      p0[v] = __expf(st0[v] - mnew);
      p1[v] = __expf(st1[v] - mnew);
      csum += p0[v] + p1[v];
    }
    csum += __shfl_xor(csum, 16, 32);
    lrun = lrun * f + csum;
    mrun = mnew;
    for (int v = 0; v < 8; ++v) { o0[v] *= f; o1[v] *= f; o2[v] *= f; o3[v] *= f; }

    // --- repack exp(S^T) into B-fragment (lane = query col, K = 32 keys) ---
    v16bf pb;
    for (int v = 0; v < 8; ++v) {
      float r0 = __shfl_xor(p0[v], 16, 32);
      float r1 = __shfl_xor(p1[v], 16, 32);
      float lo = p0[v];   // keys lh*8+0..7
      float hi = p1[v];   // keys 16+lh*8+0..7
      pb[v] = (bf16_t)lo;
      pb[8 + v] = (bf16_t)hi;
    }

    // --- O^T += V^T(dims x 32 keys) @ P^T ---
    {
      const bf16_t* vr = vb + (size_t)(0 * 16 + ln) * Sall + s;
      FragB av; av.u[0] = *(const uint4*)(vr + lh * 8);
      av.u[1] = *(const uint4*)(vr + lh * 8 + 16);
      o0 = wmma_bf16(av.v, pb, o0);
    }
    {
      const bf16_t* vr = vb + (size_t)(1 * 16 + ln) * Sall + s;
      FragB av; av.u[0] = *(const uint4*)(vr + lh * 8);
      av.u[1] = *(const uint4*)(vr + lh * 8 + 16);
      o1 = wmma_bf16(av.v, pb, o1);
    }
    {
      const bf16_t* vr = vb + (size_t)(2 * 16 + ln) * Sall + s;
      FragB av; av.u[0] = *(const uint4*)(vr + lh * 8);
      av.u[1] = *(const uint4*)(vr + lh * 8 + 16);
      o2 = wmma_bf16(av.v, pb, o2);
    }
    {
      const bf16_t* vr = vb + (size_t)(3 * 16 + ln) * Sall + s;
      FragB av; av.u[0] = *(const uint4*)(vr + lh * 8);
      av.u[1] = *(const uint4*)(vr + lh * 8 + 16);
      o3 = wmma_bf16(av.v, pb, o3);
    }
  }

  const float inv = 1.0f / lrun;
  const int t = qt * 16 + ln;       // lane owns query column
  bf16_t* orow = attn + ((size_t)(b * Tc + t)) * DIMc + h * 64;
  for (int v = 0; v < 8; ++v) {
    orow[0 * 16 + lh * 8 + v] = (bf16_t)(o0[v] * inv);
    orow[1 * 16 + lh * 8 + v] = (bf16_t)(o1[v] * inv);
    orow[2 * 16 + lh * 8 + v] = (bf16_t)(o2[v] * inv);
    orow[3 * 16 + lh * 8 + v] = (bf16_t)(o3[v] * inv);
  }
}

// ---------------------------------------------------------------------------
// y = Oraw + bo + x ; LayerNorm(y) -> bf16 (one block per row of 1024)
// ---------------------------------------------------------------------------
__global__ __launch_bounds__(256) void ln_kernel(const float* __restrict__ Oraw,
                                                 const float* __restrict__ bo,
                                                 const float* __restrict__ x,
                                                 const float* __restrict__ lng,
                                                 const float* __restrict__ lnb,
                                                 bf16_t* __restrict__ yn) {
  __shared__ float red[256];
  const int r = blockIdx.x, tid = threadIdx.x;
  float vv[4], s1 = 0.0f, s2 = 0.0f;
  for (int i = 0; i < 4; ++i) {
    int c = tid + i * 256;
    float v = Oraw[(size_t)r * DIMc + c] + bo[c] + x[(size_t)r * DIMc + c];
    vv[i] = v; s1 += v; s2 += v * v;
  }
  red[tid] = s1; __syncthreads();
  for (int o = 128; o > 0; o >>= 1) { if (tid < o) red[tid] += red[tid + o]; __syncthreads(); }
  float tot1 = red[0]; __syncthreads();
  red[tid] = s2; __syncthreads();
  for (int o = 128; o > 0; o >>= 1) { if (tid < o) red[tid] += red[tid + o]; __syncthreads(); }
  float tot2 = red[0];
  float mu = tot1 * (1.0f / DIMc);
  float var = tot2 * (1.0f / DIMc) - mu * mu;
  float rstd = rsqrtf(var + 1e-5f);
  for (int i = 0; i < 4; ++i) {
    int c = tid + i * 256;
    yn[(size_t)r * DIMc + c] = (bf16_t)((vv[i] - mu) * rstd * lng[c] + lnb[c]);
  }
}

// out = gelu_exact(Fraw + bf)
__global__ __launch_bounds__(256) void gelu_kernel(const float* __restrict__ Fraw,
                                                   const float* __restrict__ bf,
                                                   float* __restrict__ out, int n) {
  int id = blockIdx.x * 256 + threadIdx.x;
  if (id >= n) return;
  int c = id & (DIMc - 1);
  float v = Fraw[id] + bf[c];
  out[id] = 0.5f * v * (1.0f + erff(v * 0.70710678118654752f));
}

// ---------------------------------------------------------------------------
// Host orchestration
// ---------------------------------------------------------------------------
extern "C" void kernel_launch(void* const* d_in, const int* in_sizes, int n_in,
                              void* d_out, int out_size, void* d_ws, size_t ws_size,
                              hipStream_t stream) {
  const float* x    = (const float*)d_in[0];
  const float* h_a  = (const float*)d_in[1];
  const float* h_t  = (const float*)d_in[2];
  const float* p    = (const float*)d_in[3];
  const float* Wq   = (const float*)d_in[4];   const float* bq  = (const float*)d_in[5];
  const float* Wks  = (const float*)d_in[6];   const float* bks = (const float*)d_in[7];
  const float* Wvs  = (const float*)d_in[8];   const float* bvs = (const float*)d_in[9];
  const float* Wka  = (const float*)d_in[10];  const float* bka = (const float*)d_in[11];
  const float* Wva  = (const float*)d_in[12];  const float* bva = (const float*)d_in[13];
  const float* Wkt  = (const float*)d_in[14];  const float* bkt = (const float*)d_in[15];
  const float* Wvt  = (const float*)d_in[16];  const float* bvt = (const float*)d_in[17];
  const float* Wo   = (const float*)d_in[18];  const float* bo  = (const float*)d_in[19];
  const float* lng  = (const float*)d_in[20];  const float* lnb = (const float*)d_in[21];
  const float* Wf   = (const float*)d_in[22];  const float* bf  = (const float*)d_in[23];
  const float* g    = (const float*)d_in[24];
  float* out = (float*)d_out;

  // Workspace bump allocator (256B aligned)
  char* base = (char*)d_ws;
  size_t off = 0;
  auto alloc = [&](size_t bytes) -> char* {
    char* r = base + off;
    off = (off + bytes + 255) & ~(size_t)255;
    return r;
  };
  const size_t M  = (size_t)Bc * Tc;        // 4096
  bf16_t* x_bf  = (bf16_t*)alloc(M * DIMc * 2);
  bf16_t* ha_bf = (bf16_t*)alloc((size_t)Bc * KAc * SRCc * 2);
  bf16_t* ht_bf = (bf16_t*)alloc((size_t)Bc * KTc * SRCc * 2);
  bf16_t* Wq_t  = (bf16_t*)alloc((size_t)DIMc * DIMc * 2);
  bf16_t* Wks_t = (bf16_t*)alloc((size_t)DIMc * DIMc * 2);
  bf16_t* Wvs_t = (bf16_t*)alloc((size_t)DIMc * DIMc * 2);
  bf16_t* Wo_t  = (bf16_t*)alloc((size_t)DIMc * DIMc * 2);
  bf16_t* Wf_t  = (bf16_t*)alloc((size_t)DIMc * DIMc * 2);
  bf16_t* Wka_t = (bf16_t*)alloc((size_t)DIMc * SRCc * 2);
  bf16_t* Wva_t = (bf16_t*)alloc((size_t)DIMc * SRCc * 2);
  bf16_t* Wkt_t = (bf16_t*)alloc((size_t)DIMc * SRCc * 2);
  bf16_t* Wvt_t = (bf16_t*)alloc((size_t)DIMc * SRCc * 2);
  bf16_t* qh    = (bf16_t*)alloc(M * DIMc * 2);
  bf16_t* kall  = (bf16_t*)alloc((size_t)Bc * Hc * Sall * 64 * 2);
  bf16_t* vT    = (bf16_t*)alloc((size_t)Bc * Hc * Sall * 64 * 2);
  bf16_t* attn  = (bf16_t*)alloc(M * DIMc * 2);
  bf16_t* yn    = (bf16_t*)alloc(M * DIMc * 2);
  float*  raw   = (float*)alloc(M * DIMc * 4);   // reused for every GEMM output
  (void)ws_size; (void)n_in; (void)in_sizes; (void)out_size;

  const dim3 blk(256);
  // ---- convert activations ----
  cvt_bf16_kernel<<<(int)(M * DIMc / 256), blk, 0, stream>>>(x, x_bf, (int)(M * DIMc));
  build_adapter_kernel<<<Bc * KAc * SRCc / 256, blk, 0, stream>>>(h_a, p, ha_bf);
  cvt_bf16_kernel<<<Bc * KTc * SRCc / 256, blk, 0, stream>>>(h_t, ht_bf,
                                                             Bc * KTc * SRCc);
  // ---- transpose weights to (N,K) bf16 ----
  {
    dim3 tb(32, 8);
    dim3 g1(DIMc / 32, DIMc / 32);     // K=1024
    dim3 g2(DIMc / 32, SRCc / 32);     // K=2048
    transpose_cvt_kernel<<<g1, tb, 0, stream>>>(Wq,  Wq_t,  DIMc, DIMc);
    transpose_cvt_kernel<<<g1, tb, 0, stream>>>(Wks, Wks_t, DIMc, DIMc);
    transpose_cvt_kernel<<<g1, tb, 0, stream>>>(Wvs, Wvs_t, DIMc, DIMc);
    transpose_cvt_kernel<<<g1, tb, 0, stream>>>(Wo,  Wo_t,  DIMc, DIMc);
    transpose_cvt_kernel<<<g1, tb, 0, stream>>>(Wf,  Wf_t,  DIMc, DIMc);
    transpose_cvt_kernel<<<g2, tb, 0, stream>>>(Wka, Wka_t, SRCc, DIMc);
    transpose_cvt_kernel<<<g2, tb, 0, stream>>>(Wva, Wva_t, SRCc, DIMc);
    transpose_cvt_kernel<<<g2, tb, 0, stream>>>(Wkt, Wkt_t, SRCc, DIMc);
    transpose_cvt_kernel<<<g2, tb, 0, stream>>>(Wvt, Wvt_t, SRCc, DIMc);
  }

  auto gemm = [&](const bf16_t* A, const bf16_t* Wt, float* C, int m, int n, int k) {
    dim3 gg(n / 64, (m + 127) / 128);
    gemm_bf16_wmma_kernel<<<gg, blk, 0, stream>>>(A, Wt, C, m, n, k);
  };
  auto arrange = [&](const float* C, const float* bias, bf16_t* dst, int rpb,
                     int Sdst, int s_off, int transposed, int rope, float sc,
                     const float* gp) {
    int ntot = Bc * rpb * DIMc;
    arrange_kernel<<<(ntot + 255) / 256, blk, 0, stream>>>(C, bias, dst, rpb, Sdst,
                                                           s_off, transposed, rope,
                                                           sc, gp, ntot);
  };

  // ---- projections (GEMM -> bias/RoPE/scatter), raw buffer reused ----
  gemm(x_bf, Wq_t, raw, (int)M, DIMc, DIMc);
  arrange(raw, bq, qh, Tc, Tc, 0, 0, 1, 0.125f, nullptr);           // q * 1/sqrt(64)
  gemm(x_bf, Wks_t, raw, (int)M, DIMc, DIMc);
  arrange(raw, bks, kall, Tc, Sall, 0, 0, 1, 1.0f, nullptr);
  gemm(x_bf, Wvs_t, raw, (int)M, DIMc, DIMc);
  arrange(raw, bvs, vT, Tc, Sall, 0, 1, 0, 1.0f, nullptr);
  gemm(ha_bf, Wka_t, raw, Bc * KAc, DIMc, SRCc);
  arrange(raw, bka, kall, KAc, Sall, OFF_A, 0, 1, 1.0f, nullptr);
  gemm(ha_bf, Wva_t, raw, Bc * KAc, DIMc, SRCc);
  arrange(raw, bva, vT, KAc, Sall, OFF_A, 1, 0, 1.0f, nullptr);
  gemm(ht_bf, Wkt_t, raw, Bc * KTc, DIMc, SRCc);
  arrange(raw, bkt, kall, KTc, Sall, OFF_T, 0, 1, 1.0f, g);          // * tanh(g)
  gemm(ht_bf, Wvt_t, raw, Bc * KTc, DIMc, SRCc);
  arrange(raw, bvt, vT, KTc, Sall, OFF_T, 1, 0, 1.0f, nullptr);

  // ---- flash attention: 4096 (b,h,qtile) jobs, 8 waves/block ----
  attn_kernel<<<Bc * Hc * (Tc / 16) / 8, blk, 0, stream>>>(qh, kall, vT, attn);

  // ---- output projection + residual + LayerNorm ----
  gemm(attn, Wo_t, raw, (int)M, DIMc, DIMc);
  ln_kernel<<<(int)M, blk, 0, stream>>>(raw, bo, x, lng, lnb, yn);

  // ---- FFN + exact GELU ----
  gemm(yn, Wf_t, raw, (int)M, DIMc, DIMc);
  gelu_kernel<<<(int)(M * DIMc / 256), blk, 0, stream>>>(raw, bf, out,
                                                         (int)(M * DIMc));
}
